// _NeighborhoodAttentionNd_49993419325467
// MI455X (gfx1250) — matrix operand order repaired and checked by reading, not compiled, not run on hardware
//
#include <hip/hip_runtime.h>
#include <hip/hip_bf16.h>

typedef __attribute__((ext_vector_type(16))) _Float16 v16h;
typedef __attribute__((ext_vector_type(8)))  float    v8f;
typedef __attribute__((ext_vector_type(2)))  _Float16 h2;

#define B_SZ    8
#define L_SEQ   8192
#define DIM_    512
#define H_HEADS 8
#define DH_     64
#define KS_     13
#define DIL_    2

constexpr int Mtot = B_SZ * L_SEQ;   // 65536 rows (b*L + l)
constexpr int Ntot = 3 * DIM_;       // 1536 cols: [Q 512 | K 512 | V 512]
constexpr int Ktot = DIM_;           // 512
constexpr int TM = 128, TN = 128, TK = 32;
constexpr int ASTR = 40;             // halves per LDS A row (80B: 16B-aligned, bank-friendly)
constexpr int BSTR = 40;             // halves per LDS B row (stored N-major: Bs[n][k])
constexpr int NKT = Ktot / TK;       // 16 K-steps

// ---- CDNA5 async global->LDS (ASYNCcnt-tracked DMA, no VGPR data path) ----
__device__ __forceinline__ void async_copy_b128(unsigned lds_off, const void* gaddr) {
    unsigned long long ga = (unsigned long long)(uintptr_t)gaddr;
    asm volatile("global_load_async_to_lds_b128 %0, %1, off"
                 :: "v"(lds_off), "v"(ga) : "memory");
}
__device__ __forceinline__ void wait_async0() {
    asm volatile("s_wait_asynccnt 0x0" ::: "memory");
}

// ---------------------------------------------------------------------------
// Kernel 1: pack Wq|Wk|Wv into f16, transposed to [N=1536][K=512] for coalesced
// B-tile staging in the GEMM.
// ---------------------------------------------------------------------------
__global__ __launch_bounds__(256)
void wt_convert_kernel(const float* __restrict__ Wq, const float* __restrict__ Wk,
                       const float* __restrict__ Wv, _Float16* __restrict__ Wt)
{
    int idx = blockIdx.x * 256 + threadIdx.x;
    if (idx >= Ntot * Ktot) return;
    int n = idx >> 9;                 // 0..1535
    int k = idx & (Ktot - 1);         // 0..511
    const float* src = (n < DIM_) ? Wq : ((n < 2 * DIM_) ? Wk : Wv);
    Wt[idx] = (_Float16)src[k * DIM_ + (n & (DIM_ - 1))];
}

// ---------------------------------------------------------------------------
// Kernel 2: fused QKV GEMM, f16 WMMA with f32 accumulate.
//   C[m][n] = sum_k X[m][k] * W[k][n] + bias[n],  stored as f16 into qkv.
// Block tile 128x128, K-step 32, 8 waves (4M x 2N), each wave 2x4 WMMA tiles.
// A tile: f32 global -> regs -> cvt -> LDS (needs conversion).
// B tile: f16 global -> LDS via global_load_async_to_lds_b128 (ASYNCcnt).
// ---------------------------------------------------------------------------
union FragU { uint4 u[2]; v16h v; };

__global__ __launch_bounds__(256)
void qkv_gemm_kernel(const float* __restrict__ X, const _Float16* __restrict__ Wt,
                     const float* __restrict__ bq, const float* __restrict__ bk,
                     const float* __restrict__ bv, _Float16* __restrict__ qkv)
{
    __shared__ _Float16 As[2][TM * ASTR];   // As[buf][m*ASTR + k]
    __shared__ _Float16 Bs[2][TN * BSTR];   // Bs[buf][n*BSTR + k]  (N-major!)

    const int tid   = threadIdx.x;
    const int lane  = tid & 31;
    const int wid   = tid >> 5;
    const int tlo   = lane & 15;      // row/col-in-tile lane id
    const int thi   = lane >> 4;      // half-wave select
    const int waveM = wid & 3;        // 4 waves along M (32 rows each)
    const int waveN = wid >> 2;       // 2 waves along N (64 cols each)

    const int    n0 = blockIdx.x * TN;                 // N-tiles fastest -> share X via L2
    const size_t m0 = (size_t)blockIdx.y * TM;

    const int srow = tid >> 1;            // 0..127 : staging row
    const int scol = (tid & 1) * 16;      // 0 or 16 : staging k-subrange

    // per-thread staging addresses
    const float*    gA = X  + (m0 + srow) * Ktot + scol;
    const _Float16* gB = Wt + (size_t)(n0 + srow) * Ktot + scol;
    const unsigned  ldsB0 = (unsigned)(uintptr_t)&Bs[0][srow * BSTR + scol];
    const unsigned  ldsB1 = (unsigned)(uintptr_t)&Bs[1][srow * BSTR + scol];

    v8f acc[2][4];
#pragma unroll
    for (int mi = 0; mi < 2; ++mi)
#pragma unroll
        for (int ni = 0; ni < 4; ++ni)
            acc[mi][ni] = (v8f){0.f, 0.f, 0.f, 0.f, 0.f, 0.f, 0.f, 0.f};

    float4 ar[4];     // 16 f32 of X (converted to f16 at LDS store)

    // ---- stage tile 0 ----
    {
        // B: async DMA straight into LDS
        async_copy_b128(ldsB0,      gB);
        async_copy_b128(ldsB0 + 16, gB + 8);
        // A: load f32, convert, store f16
        ar[0] = *(const float4*)(gA + 0);
        ar[1] = *(const float4*)(gA + 4);
        ar[2] = *(const float4*)(gA + 8);
        ar[3] = *(const float4*)(gA + 12);
        _Float16 tmp[16];
#pragma unroll
        for (int i = 0; i < 4; ++i) {
            tmp[4*i+0] = (_Float16)((&ar[i].x)[0]);
            tmp[4*i+1] = (_Float16)((&ar[i].x)[1]);
            tmp[4*i+2] = (_Float16)((&ar[i].x)[2]);
            tmp[4*i+3] = (_Float16)((&ar[i].x)[3]);
        }
        _Float16* da = &As[0][srow * ASTR + scol];
        *(uint4*)(da + 0) = *(const uint4*)(tmp + 0);
        *(uint4*)(da + 8) = *(const uint4*)(tmp + 8);
    }
    wait_async0();
    __syncthreads();

    int buf = 0;
    for (int kt = 0; kt < NKT; ++kt) {
        // prefetch next tile: B via async DMA into the idle buffer
        // (previous readers of buf^1 finished at the barrier that ended iter kt-1)
        if (kt + 1 < NKT) {
            const _Float16* nb = gB + (kt + 1) * TK;
            const unsigned  lb = (buf == 0) ? ldsB1 : ldsB0;
            async_copy_b128(lb,      nb);
            async_copy_b128(lb + 16, nb + 8);
            const float* na = gA + (kt + 1) * TK;
            ar[0] = *(const float4*)(na + 0);
            ar[1] = *(const float4*)(na + 4);
            ar[2] = *(const float4*)(na + 8);
            ar[3] = *(const float4*)(na + 12);
        }

        // ---- fragment loads per ISA layout ----
        // A 16x32 f16: lane t -> row t&15, K = (t>>4)*8 + {0..7} and 16+{0..7}
        v16h afr[2];
#pragma unroll
        for (int mi = 0; mi < 2; ++mi) {
            const _Float16* pa = &As[buf][(waveM * 32 + mi * 16 + tlo) * ASTR];
            FragU f;
            f.u[0] = *(const uint4*)(pa + thi * 8);
            f.u[1] = *(const uint4*)(pa + 16 + thi * 8);
            afr[mi] = f.v;
        }
        // B 32x16 f16: lane t -> col t&15, K = (t>>4)*8 + {0..7} and 16+{0..7}
        v16h bfr[4];
#pragma unroll
        for (int ni = 0; ni < 4; ++ni) {
            const _Float16* pb = &Bs[buf][(waveN * 64 + ni * 16 + tlo) * BSTR];
            FragU f;
            f.u[0] = *(const uint4*)(pb + thi * 8);
            f.u[1] = *(const uint4*)(pb + 16 + thi * 8);
            bfr[ni] = f.v;
        }

#pragma unroll
        for (int mi = 0; mi < 2; ++mi)
#pragma unroll
            for (int ni = 0; ni < 4; ++ni)
                acc[mi][ni] = __builtin_amdgcn_wmma_f32_16x16x32_f16(
                    false, afr[mi], false, bfr[ni],
                    (short)0, acc[mi][ni], false, false);

        // store prefetched A into the other buffer
        if (kt + 1 < NKT) {
            _Float16 tmp[16];
#pragma unroll
            for (int i = 0; i < 4; ++i) {
                tmp[4*i+0] = (_Float16)((&ar[i].x)[0]);
                tmp[4*i+1] = (_Float16)((&ar[i].x)[1]);
                tmp[4*i+2] = (_Float16)((&ar[i].x)[2]);
                tmp[4*i+3] = (_Float16)((&ar[i].x)[3]);
            }
            _Float16* da = &As[buf ^ 1][srow * ASTR + scol];
            *(uint4*)(da + 0) = *(const uint4*)(tmp + 0);
            *(uint4*)(da + 8) = *(const uint4*)(tmp + 8);
        }
        wait_async0();       // B DMA for buf^1 landed
        __syncthreads();
        buf ^= 1;
    }

    // ---- epilogue: C/D layout -> lane t owns col (t&15), rows (t>>4)*8 + r ----
#pragma unroll
    for (int ni = 0; ni < 4; ++ni) {
        const int col = n0 + waveN * 64 + ni * 16 + tlo;
        const float bias = (col < DIM_) ? bq[col]
                         : (col < 2 * DIM_) ? bk[col - DIM_] : bv[col - 2 * DIM_];
#pragma unroll
        for (int mi = 0; mi < 2; ++mi) {
            const size_t rbase = m0 + (size_t)(waveM * 32 + mi * 16 + thi * 8);
            v8f c = acc[mi][ni];
#pragma unroll
            for (int r = 0; r < 8; ++r)
                qkv[(rbase + r) * Ntot + col] = (_Float16)(c[r] + bias);
        }
    }
}

// ---------------------------------------------------------------------------
// Kernel 3: 1D neighborhood attention (KS=13, DIL=2) with RPB + softmax.
// One wave32 per query (b,h,l); lane owns 2 of the 64 head dims.
// ---------------------------------------------------------------------------
__global__ __launch_bounds__(256)
void na1d_kernel(const _Float16* __restrict__ qkv, const float* __restrict__ rpb,
                 float* __restrict__ out)
{
    const int lane = threadIdx.x & 31;
    const int wid  = threadIdx.x >> 5;
    const long long q = (long long)blockIdx.x * 8 + wid;   // ((b*H + h)*L + l)
    const int l  = (int)(q & (L_SEQ - 1));
    const int bh = (int)(q >> 13);
    const int h  = bh & (H_HEADS - 1);
    const int b  = bh >> 3;

    // NATTEN window-start / rpb-start (dilation > 1 path; bb == 0 for L=8192)
    const int ns    = KS_ / 2;                         // 6
    const int imodd = l & (DIL_ - 1);
    const int aa    = (L_SEQ / DIL_) * DIL_;
    const int bb    = L_SEQ - aa;
    const int hi    = (imodd < bb) ? (L_SEQ - bb + imodd - 2 * ns * DIL_)
                                   : (aa + imodd - KS_ * DIL_);
    const int  niq   = l - ns * DIL_;
    const bool at_hi = (l + ns * DIL_) >= L_SEQ;
    const int  ws = (niq < 0) ? imodd : (at_hi ? hi : niq);
    const int  ps = (niq < 0) ? (KS_ - 1 - l / DIL_)
                              : (at_hi ? ((L_SEQ - l - 1) / DIL_) : ns);

    const size_t rowbase = (size_t)b * L_SEQ * (size_t)Ntot;
    const int    coff    = h * DH_ + 2 * lane;

    h2 qv = *(const h2*)(qkv + rowbase + (size_t)l * Ntot + coff);
    const float qs = 0.125f;                           // 1/sqrt(DH)
    const float q0 = (float)qv.x * qs, q1 = (float)qv.y * qs;

    float s[KS_];
#pragma unroll
    for (int j = 0; j < KS_; ++j) {
        const int idx = ws + j * DIL_;
        h2 kv = *(const h2*)(qkv + rowbase + (size_t)idx * Ntot + DIM_ + coff);
        float p = q0 * (float)kv.x + q1 * (float)kv.y;
#pragma unroll
        for (int off = 16; off > 0; off >>= 1) p += __shfl_xor(p, off, 32);
        s[j] = p + rpb[h * (2 * KS_ - 1) + ps + j];
    }

    float mx = s[0];
#pragma unroll
    for (int j = 1; j < KS_; ++j) mx = fmaxf(mx, s[j]);
    float den = 0.f;
#pragma unroll
    for (int j = 0; j < KS_; ++j) { s[j] = __expf(s[j] - mx); den += s[j]; }
    const float inv = 1.0f / den;

    float o0 = 0.f, o1 = 0.f;
#pragma unroll
    for (int j = 0; j < KS_; ++j) {
        const int idx = ws + j * DIL_;
        h2 vv = *(const h2*)(qkv + rowbase + (size_t)idx * Ntot + 2 * DIM_ + coff);
        const float w = s[j] * inv;
        o0 += w * (float)vv.x;
        o1 += w * (float)vv.y;
    }
    float2 o = make_float2(o0, o1);
    *(float2*)(out + ((size_t)b * L_SEQ + l) * DIM_ + coff) = o;
}

// ---------------------------------------------------------------------------
extern "C" void kernel_launch(void* const* d_in, const int* in_sizes, int n_in,
                              void* d_out, int out_size, void* d_ws, size_t ws_size,
                              hipStream_t stream)
{
    const float* X   = (const float*)d_in[0];
    const float* Wq  = (const float*)d_in[1];
    const float* bq  = (const float*)d_in[2];
    const float* Wk  = (const float*)d_in[3];
    const float* bk  = (const float*)d_in[4];
    const float* Wv  = (const float*)d_in[5];
    const float* bv  = (const float*)d_in[6];
    const float* rpb = (const float*)d_in[7];
    float* out = (float*)d_out;

    // workspace layout: [Wt f16 1536x512][qkv f16 65536x1536]
    _Float16* Wt  = (_Float16*)d_ws;
    _Float16* qkv = (_Float16*)((char*)d_ws + (size_t)Ntot * Ktot * sizeof(_Float16));

    wt_convert_kernel<<<(Ntot * Ktot + 255) / 256, 256, 0, stream>>>(Wq, Wk, Wv, Wt);

    dim3 grid(Ntot / TN, Mtot / TM);   // N-tiles fastest: co-resident blocks share X rows in L2
    qkv_gemm_kernel<<<grid, 256, 0, stream>>>(X, Wt, bq, bk, bv, qkv);

    na1d_kernel<<<(B_SZ * H_HEADS * L_SEQ) / 8, 256, 0, stream>>>(qkv, rpb, out);
}
